// eGRU_86105504350683
// MI455X (gfx1250) — matrix operand order repaired and checked by reading, not compiled, not run on hardware
//
// CDNA5 / gfx1250 GRU (T=2048, B=128, I=H=256)
//
// Phase 1 (k_gi):   gi = x @ w_ih^T + b_ih   -- parallel bf16-WMMA GEMM.
//                   x read once from HBM (f32->bf16 in-register); w_ih slice
//                   staged per 64-column group into LDS so the 8 waves of a
//                   block share one copy (ds_load_b128 B-fragments).
// Phase 2 (k_scan): 16 persistent workgroups, each owning 16 h-columns
//                   (gate columns c, 256+c, 512+c). w_hh slice lives in LDS
//                   (bf16); h state double-buffered in ws (bf16) as the WMMA
//                   A-operand; the wave-private h slice is carried in
//                   registers. gi strip for step t is loaded up-front to
//                   overlap the WMMAs; step t+1's strip is prefetched
//                   (global_prefetch_b8) before the device-wide barrier
//                   (monotonic atomic counter + s_sleep spin).
//
// All matrix math uses v_wmma_f32_16x16x32_bf16 (f32 accumulate).

#include <hip/hip_runtime.h>
#include <hip/hip_bf16.h>

#define T_STEPS 2048
#define BATCH   128
#define IDIM    256
#define HDIM    256
#define GDIM    768   // 3*H
#define NWG_SCAN 16

typedef __bf16 bf16;
typedef bf16  v16bf __attribute__((ext_vector_type(16)));
typedef float v8f   __attribute__((ext_vector_type(8)));

union FragB16 {
    v16bf    v;
    uint4    q[2];
    unsigned u[8];
};

__device__ __forceinline__ unsigned pack2bf(float lo, float hi) {
    union { bf16 h[2]; unsigned u; } p;
    p.h[0] = (bf16)lo;
    p.h[1] = (bf16)hi;
    return p.u;
}

__device__ __forceinline__ float sigmoidf_(float x) {
    return 1.0f / (1.0f + __expf(-x));
}

__device__ __forceinline__ float tanhf_(float x) {
    float e = __expf(-2.0f * fabsf(x));
    float t = (1.0f - e) / (1.0f + e);
    return copysignf(t, x);
}

// ---------------------------------------------------------------------------
// Kernel 0: convert w_ih -> bf16, h0 -> bf16 (scan h-buffer 0), reset barrier.
// Runs every launch (graph-replay safe re-initialization).
// ---------------------------------------------------------------------------
__global__ void k_prep(const float* __restrict__ w_ih,
                       const float* __restrict__ h0,
                       bf16* __restrict__ wih_bf,
                       bf16* __restrict__ hbf0,
                       unsigned* __restrict__ bar) {
    int i = blockIdx.x * blockDim.x + threadIdx.x;
    int stride = gridDim.x * blockDim.x;
    for (int j = i; j < GDIM * IDIM; j += stride) wih_bf[j] = (bf16)w_ih[j];
    for (int j = i; j < BATCH * HDIM; j += stride) hbf0[j] = (bf16)h0[j];
    if (i == 0) *bar = 0u;
}

// ---------------------------------------------------------------------------
// Kernel 1: gi[t*B+b, g] = sum_i x[t,b,i]*w_ih[g,i] + b_ih[g]
// Grid: 2048 blocks x 256 threads (8 waves). Each wave owns a 16-row M tile,
// preloads its 16x256 A panel (f32 -> bf16 fragments) once. The 768-wide N
// dimension is swept in 64-column groups whose w_ih slice (32 KB bf16) is
// staged in LDS and shared by all 8 waves.
// ---------------------------------------------------------------------------
__global__ __launch_bounds__(256) void k_gi(const float* __restrict__ x,
                                            const bf16* __restrict__ wih_bf,
                                            const float* __restrict__ b_ih,
                                            float* __restrict__ gi) {
    __shared__ bf16 sB[64 * 264];   // 64 g-rows x 256 K (pad 8)

    const int tid  = threadIdx.x;
    const int lane = tid & 31;
    const int wave = tid >> 5;
    const int lr   = lane & 15;   // A row / B column within tile
    const int grp  = lane >> 4;   // K-half selector
    const size_t rowBase = (size_t)blockIdx.x * 128 + (size_t)wave * 16;

    // Preload A fragments: 8 K-steps of 32, per ISA 16-bit A layout.
    FragB16 A[8];
    const float* xrow = x + (rowBase + (size_t)lr) * IDIM;
#pragma unroll
    for (int k = 0; k < 8; ++k) {
        const int kb = k * 32 + grp * 8;
        const float4* p0 = (const float4*)(xrow + kb);        // K = kb..kb+7
        const float4* p1 = (const float4*)(xrow + kb + 16);   // K = kb+16..kb+23
        float4 f0 = p0[0], f1 = p0[1], f2 = p1[0], f3 = p1[1];
        A[k].u[0] = pack2bf(f0.x, f0.y); A[k].u[1] = pack2bf(f0.z, f0.w);
        A[k].u[2] = pack2bf(f1.x, f1.y); A[k].u[3] = pack2bf(f1.z, f1.w);
        A[k].u[4] = pack2bf(f2.x, f2.y); A[k].u[5] = pack2bf(f2.z, f2.w);
        A[k].u[6] = pack2bf(f3.x, f3.y); A[k].u[7] = pack2bf(f3.z, f3.w);
    }

    const v8f vz = {0.f, 0.f, 0.f, 0.f, 0.f, 0.f, 0.f, 0.f};

    for (int ng = 0; ng < 12; ++ng) {
        const int n0 = ng * 64;

        __syncthreads();   // sB reuse: previous group's readers are done
        // Stage w_ih rows [n0, n0+64) into LDS: 2048 16-byte chunks.
        for (int c = tid; c < 2048; c += 256) {
            const int row = c >> 5;
            const int ke  = (c & 31) * 8;
            *(uint4*)&sB[row * 264 + ke] =
                *(const uint4*)(wih_bf + (size_t)(n0 + row) * IDIM + ke);
        }
        __syncthreads();

        v8f acc[4];
#pragma unroll
        for (int nt = 0; nt < 4; ++nt) acc[nt] = vz;

        for (int k = 0; k < 8; ++k) {
#pragma unroll
            for (int nt = 0; nt < 4; ++nt) {
                // B(k,n) = w_ih[n0+nt*16+n, k] : lane n holds K = kb..kb+7, kb+16..kb+23.
                const bf16* bp = &sB[(nt * 16 + lr) * 264 + k * 32 + grp * 8];
                FragB16 B;
                B.q[0] = *(const uint4*)bp;
                B.q[1] = *(const uint4*)(bp + 16);
                acc[nt] = __builtin_amdgcn_wmma_f32_16x16x32_bf16(
                    false, A[k].v, false, B.v, (short)0, acc[nt], false, false);
            }
        }
#pragma unroll
        for (int nt = 0; nt < 4; ++nt) {
            const int g = n0 + nt * 16 + lr;
            const float bias = b_ih[g];
#pragma unroll
            for (int r = 0; r < 8; ++r) {
                const size_t row = rowBase + r + grp * 8;
                gi[row * GDIM + g] = acc[nt][r] + bias;
            }
        }
    }
}

// ---------------------------------------------------------------------------
// Kernel 2: sequential GRU scan. 16 persistent workgroups x 256 threads.
// Block blk owns h-columns [blk*16, blk*16+16) and gate rows of w_hh at
// {c, 256+c, 512+c}. Wave w owns batch rows [w*16, w*16+16).
// ---------------------------------------------------------------------------
__global__ __launch_bounds__(256) void k_scan(const float* __restrict__ gi,
                                              const float* __restrict__ w_hh,
                                              const float* __restrict__ b_hh,
                                              const float* __restrict__ h0,
                                              bf16* __restrict__ hbuf0,
                                              bf16* __restrict__ hbuf1,
                                              unsigned* __restrict__ bar,
                                              float* __restrict__ outs,
                                              float* __restrict__ hlast) {
    __shared__ bf16 sW[48 * 264];   // 3 gates x 16 rows x 256 K (pad 8)

    const int tid  = threadIdx.x;
    const int lane = tid & 31;
    const int wave = tid >> 5;
    const int lr   = lane & 15;
    const int grp  = lane >> 4;
    const int blk  = blockIdx.x;
    const int c0   = blk * 16;      // h-column base
    const int m0   = wave * 16;     // batch-row base

    // Stage w_hh slice (f32 -> bf16) into LDS: local row s*16+j <- global row
    // s*256 + c0 + j, 4 K-elements per chunk.
    for (int c = tid; c < 48 * 64; c += 256) {
        const int row  = c >> 6;
        const int k4   = (c & 63) << 2;
        const int grow = (row >> 4) * HDIM + c0 + (row & 15);
        float4 f = *(const float4*)(w_hh + (size_t)grow * HDIM + k4);
        *(uint2*)&sW[row * 264 + k4] =
            make_uint2(pack2bf(f.x, f.y), pack2bf(f.z, f.w));
    }
    __syncthreads();

    // Per-lane invariants: this lane always handles column c0+lr.
    const float bhr = b_hh[c0 + lr];
    const float bhz = b_hh[256 + c0 + lr];
    const float bhn = b_hh[512 + c0 + lr];

    float hprev[8];
#pragma unroll
    for (int r = 0; r < 8; ++r)
        hprev[r] = h0[(size_t)(m0 + r + grp * 8) * HDIM + c0 + lr];

    const v8f vz = {0.f, 0.f, 0.f, 0.f, 0.f, 0.f, 0.f, 0.f};

    // Prefetch step 0's gi strip (written by k_gi, no sync needed).
    for (int idx = tid; idx < 384; idx += 256) {
        const int gate = idx >> 7;          // 0..2
        const int row  = idx & 127;         // batch row
        __builtin_prefetch(gi + (size_t)row * GDIM + gate * 256 + c0, 0, 1);
    }

    for (int t = 0; t < T_STEPS; ++t) {
        const bf16* hc = (t & 1) ? hbuf1 : hbuf0;   // read
        bf16*       hn = (t & 1) ? hbuf0 : hbuf1;   // write (next step's A)

        // Issue this step's gi loads first so they overlap the WMMAs.
        const size_t gib = (size_t)t * BATCH * GDIM;
        float giR[8], giZ[8], giN[8];
#pragma unroll
        for (int r = 0; r < 8; ++r) {
            const int b = m0 + r + grp * 8;
            const size_t gb = gib + (size_t)b * GDIM;
            giR[r] = gi[gb + c0 + lr];
            giZ[r] = gi[gb + 256 + c0 + lr];
            giN[r] = gi[gb + 512 + c0 + lr];
        }

        v8f accR = vz, accZ = vz, accN = vz;
#pragma unroll
        for (int k = 0; k < 8; ++k) {
            const int kb = k * 32;
            const bf16* ap = hc + (size_t)(m0 + lr) * HDIM + kb + grp * 8;
            FragB16 A;
            A.q[0] = *(const uint4*)ap;
            A.q[1] = *(const uint4*)(ap + 16);

            const bf16* b0 = &sW[(0 * 16 + lr) * 264 + kb + grp * 8];
            const bf16* b1 = &sW[(1 * 16 + lr) * 264 + kb + grp * 8];
            const bf16* b2 = &sW[(2 * 16 + lr) * 264 + kb + grp * 8];
            FragB16 BR, BZ, BN;
            BR.q[0] = *(const uint4*)b0; BR.q[1] = *(const uint4*)(b0 + 16);
            BZ.q[0] = *(const uint4*)b1; BZ.q[1] = *(const uint4*)(b1 + 16);
            BN.q[0] = *(const uint4*)b2; BN.q[1] = *(const uint4*)(b2 + 16);

            accR = __builtin_amdgcn_wmma_f32_16x16x32_bf16(
                false, A.v, false, BR.v, (short)0, accR, false, false);
            accZ = __builtin_amdgcn_wmma_f32_16x16x32_bf16(
                false, A.v, false, BZ.v, (short)0, accZ, false, false);
            accN = __builtin_amdgcn_wmma_f32_16x16x32_bf16(
                false, A.v, false, BN.v, (short)0, accN, false, false);
        }

        // Gates + state update. C layout: VGPR r -> batch row m0+r+grp*8,
        // lane -> column c0+lr.
#pragma unroll
        for (int r = 0; r < 8; ++r) {
            const int b = m0 + r + grp * 8;
            const float rr = sigmoidf_(giR[r] + accR[r] + bhr);
            const float zz = sigmoidf_(giZ[r] + accZ[r] + bhz);
            const float nn = tanhf_(giN[r] + rr * (accN[r] + bhn));
            const float hv = (1.0f - zz) * nn + zz * hprev[r];
            hprev[r] = hv;
            outs[((size_t)t * BATCH + b) * HDIM + c0 + lr] = hv;
            hn[(size_t)b * HDIM + c0 + lr] = (bf16)hv;
        }

        // Prefetch next step's gi strip into cache while we sit at the
        // barrier (global_prefetch_b8: no counters, fire-and-forget).
        if (t + 1 < T_STEPS) {
            const size_t gibn = (size_t)(t + 1) * BATCH * GDIM;
            for (int idx = tid; idx < 384; idx += 256) {
                const int gate = idx >> 7;
                const int row  = idx & 127;
                __builtin_prefetch(
                    gi + gibn + (size_t)row * GDIM + gate * 256 + c0, 0, 1);
            }
        }

        // Device-wide split barrier: monotonic counter, one arrival per WG.
        __threadfence();
        __syncthreads();
        if (tid == 0) {
            atomicAdd(bar, 1u);
            const unsigned target = (unsigned)NWG_SCAN * (unsigned)(t + 1);
            while (atomicAdd(bar, 0u) < target) __builtin_amdgcn_s_sleep(4);
        }
        __syncthreads();
        __threadfence();   // acquire: make peers' h writes visible
    }

#pragma unroll
    for (int r = 0; r < 8; ++r) {
        const int b = m0 + r + grp * 8;
        hlast[(size_t)b * HDIM + c0 + lr] = hprev[r];
    }
}

// ---------------------------------------------------------------------------
// Launch. Inputs: x, h0, w_ih, w_hh, b_ih, b_hh (all f32).
// Output: outs [T,B,H] f32 ++ h_last [1,B,H] f32.
// Workspace: [0, 384K)  w_ih bf16
//            [384K, 448K) h buffer 0 (bf16)   [448K, 512K) h buffer 1
//            [512K]     barrier counter
//            [1M, 1M+768M) gi f32
// ---------------------------------------------------------------------------
extern "C" void kernel_launch(void* const* d_in, const int* in_sizes, int n_in,
                              void* d_out, int out_size, void* d_ws,
                              size_t ws_size, hipStream_t stream) {
    (void)in_sizes; (void)n_in; (void)out_size; (void)ws_size;

    const float* x    = (const float*)d_in[0];
    const float* h0   = (const float*)d_in[1];
    const float* w_ih = (const float*)d_in[2];
    const float* w_hh = (const float*)d_in[3];
    const float* b_ih = (const float*)d_in[4];
    const float* b_hh = (const float*)d_in[5];

    char* ws = (char*)d_ws;
    bf16*     wih_bf = (bf16*)(ws);
    bf16*     hbf0   = (bf16*)(ws + 393216);
    bf16*     hbf1   = (bf16*)(ws + 393216 + 65536);
    unsigned* bar    = (unsigned*)(ws + 524288);
    float*    gi     = (float*)(ws + 1048576);

    float* outs  = (float*)d_out;
    float* hlast = outs + (size_t)T_STEPS * BATCH * HDIM;

    k_prep<<<384, 256, 0, stream>>>(w_ih, h0, wih_bf, hbf0, bar);
    k_gi<<<2048, 256, 0, stream>>>(x, wih_bf, b_ih, gi);
    k_scan<<<NWG_SCAN, 256, 0, stream>>>(gi, w_hh, b_hh, h0, hbf0, hbf1, bar,
                                         outs, hlast);
}
